// FluidLayerVideo_6055903888137
// MI455X (gfx1250) — compile-verified, matrix operand order checked by reading, not run on hardware
//
#include <hip/hip_runtime.h>
#include <math.h>

#define B_ 2
#define C_ 128
#define T_ 8
#define H_ 64
#define W_ 64
#define N_ (T_*H_*W_)      /* 32768 */
#define BN_ (B_*N_)        /* 65536 */
#define HID_ 256
#define MAX_STEPS_ 12
#define DIFF_SCALE 0.25f
#define PROBE_CELLS (B_*C_*2*8*8)   /* 32768 */

typedef __attribute__((ext_vector_type(16))) _Float16 v16h;
typedef __attribute__((ext_vector_type(8)))  _Float16 h8;
typedef __attribute__((ext_vector_type(8)))  float    v8f;

__device__ __forceinline__ float softplus_f(float x){ return x > 20.f ? x : log1pf(expf(x)); }
__device__ __forceinline__ float gelu_f(float x){ return 0.5f*x*(1.f + erff(x*0.70710678118654752f)); }
__device__ __forceinline__ int imin_(int a, int b){ return a < b ? a : b; }

__device__ __forceinline__ float blockReduceSum(float v){
  __shared__ float red[8];
  int lane = threadIdx.x & 31, wid = threadIdx.x >> 5;
  #pragma unroll
  for (int off = 16; off > 0; off >>= 1) v += __shfl_down(v, off);
  if (lane == 0) red[wid] = v;
  __syncthreads();
  v = (threadIdx.x < (blockDim.x >> 5)) ? red[threadIdx.x] : 0.f;
  if (wid == 0){
    #pragma unroll
    for (int off = 4; off > 0; off >>= 1) v += __shfl_down(v, off);
  }
  __syncthreads();
  return v;  /* valid on thread 0 */
}

/* ---- WMMA fragment loader (CDNA5 16-bit layout, wave32) ----
 * Works for A (row-major MxK) and for B when B is stored TRANSPOSED (N-major)
 * in LDS: lane L picks row r0+(L&15); half-wave hf=L>>4 selects K groups
 * hf*8+{0..7} and 16+hf*8+{0..7} -> two contiguous 16B LDS reads (ds_load_b128).
 */
__device__ __forceinline__ v16h load_frag_row(const _Float16* base, int ld, int r0, int k0, int lane){
  int r = r0 + (lane & 15);
  int hf = lane >> 4;
  const _Float16* p = base + r*ld + k0 + hf*8;
  h8 lo = *(const h8*)p;
  h8 hi = *(const h8*)(p + 16);
  return __builtin_shufflevector(lo, hi, 0,1,2,3,4,5,6,7,8,9,10,11,12,13,14,15);
}

/* ---------------- kernels ---------------- */

__global__ void zero_kernel(float* p, int n){
  int i = blockIdx.x*blockDim.x + threadIdx.x;
  if (i < n) p[i] = 0.f;
}

__global__ void prep_kernel(const float* w1, const float* w2,
                            const float* ks, const float* kt,
                            const float* ag, const float* al, const float* logdt,
                            _Float16* w1h, _Float16* w2h,
                            float* spsks, float* spskt, float* scal){
  int i = blockIdx.x*blockDim.x + threadIdx.x;
  int stride = gridDim.x*blockDim.x;
  for (int j = i; j < C_*HID_; j += stride) w1h[j] = (_Float16)w1[j];
  for (int j = i; j < HID_*C_; j += stride) w2h[j] = (_Float16)w2[j];
  for (int j = i; j < 3*C_;   j += stride) spsks[j] = softplus_f(ks[j]);
  for (int j = i; j < 2*C_;   j += stride) spskt[j] = softplus_f(kt[j]);
  if (i == 0){
    float dt = expf(logdt[0]);
    dt = fminf(fmaxf(dt, 0.005f), 0.35f);
    scal[0] = dt;
    scal[1] = softplus_f(ag[0]);
    scal[2] = softplus_f(al[0]);
  }
}

/* generic block-mean pool over (B,C,T,H,W) channel-first -> (B,C,tp,hp,wp) */
__global__ void pool_kernel(const float* in, float* out, int tp, int hp, int wp){
  int cells = B_*C_*tp*hp*wp;
  int id = blockIdx.x*blockDim.x + threadIdx.x;
  if (id >= cells) return;
  int pw = id % wp; int tmp = id / wp;
  int ph = tmp % hp; tmp /= hp;
  int pt = tmp % tp; tmp /= tp;
  int c = tmp % C_;  int b = tmp / C_;
  int bt = T_/tp, bh = H_/hp, bw = W_/wp;
  const float* base = in + (size_t)(b*C_ + c)*N_;
  float s = 0.f;
  for (int it = 0; it < bt; ++it)
    for (int ih = 0; ih < bh; ++ih)
      for (int iw = 0; iw < bw; ++iw)
        s += base[((pt*bt+it)*H_ + (ph*bh+ih))*W_ + (pw*bw+iw)];
  out[id] = s / (float)(bt*bh*bw);
}

__global__ void pooled_mean_kernel(const float* u, float* pooled){
  int bc = blockIdx.x;  /* b*C + c */
  const float* base = u + (size_t)bc*N_;
  float s = 0.f;
  for (int i = threadIdx.x; i < N_; i += blockDim.x) s += base[i];
  s = blockReduceSum(s);
  if (threadIdx.x == 0) pooled[bc] = s / (float)N_;
}

__global__ void gate_value_kernel(const float* pooled, const float* gw, const float* gb,
                                  const float* vw, const float* vb, float* h){
  int tid = threadIdx.x;              /* 256 threads */
  int b = tid >> 7, c = tid & 127;
  const float* p = pooled + b*C_;
  float g = gb[c], v = vb[c];
  for (int k = 0; k < C_; ++k){
    float pk = p[k];
    g += pk * gw[k*C_ + c];
    v += pk * vw[k*C_ + c];
  }
  h[tid] += (1.f/(1.f+expf(-g))) * tanhf(v);
}

__global__ void local_proj_kernel(const float* lsmall, const float* wlp, const float* blp, float* lproj){
  int id = blockIdx.x*blockDim.x + threadIdx.x;   /* B*C*32 = 8192 */
  if (id >= B_*C_*32) return;
  int cell = id & 31;
  int oc = (id >> 5) & 127;
  int b = id >> 12;
  float s = blp[oc];
  for (int c = 0; c < C_; ++c)
    s += lsmall[((size_t)(b*C_ + c) << 5) + cell] * wlp[c*C_ + oc];  /* einsum with W.T */
  lproj[((size_t)(b*C_ + oc) << 5) + cell] = s;
}

/* fused MLP: react = GELU(u@W1 + b1) @ W2 + b2, f16 WMMA, hidden stays in LDS */
__global__ void __launch_bounds__(256)
reaction_mlp_kernel(const float* u, const _Float16* w1h, const float* b1,
                    const _Float16* w2h, const float* b2, float* react){
  __shared__ _Float16 As[128][128];   /* 32 KB: A tile, rows=n, cols=c              */
  __shared__ _Float16 Bt[32][128];    /*  8 KB: W1 chunk TRANSPOSED (Nh x K)        */
  __shared__ _Float16 Hs[128][32];    /*  8 KB: hidden chunk (M x Kh)               */
  __shared__ _Float16 W2t[128][32];   /*  8 KB: W2 chunk TRANSPOSED (N x Kh)        */

  const int tid  = threadIdx.x;
  const int lane = tid & 31;
  const int wv   = tid >> 5;      /* 0..7 */
  const int m0   = wv * 16;
  const int hf   = lane >> 4;
  const int col  = lane & 15;

  const int rowBase = blockIdx.x * 128;
  const int b  = rowBase / N_;
  const int n0 = rowBase - b * N_;
  const float* ub = u + (size_t)b * C_ * N_;

  /* stage A tile: float4 global reads (coalesced along n), f16 LDS writes */
  for (int idx = tid; idx < 128*32; idx += 256){
    int c = idx >> 5;
    int q = idx & 31;
    const float4 v = *(const float4*)(ub + (size_t)c * N_ + n0 + q*4);
    As[q*4+0][c] = (_Float16)v.x;
    As[q*4+1][c] = (_Float16)v.y;
    As[q*4+2][c] = (_Float16)v.z;
    As[q*4+3][c] = (_Float16)v.w;
  }

  const v8f z8 = {0.f,0.f,0.f,0.f,0.f,0.f,0.f,0.f};
  v8f accC[8];
  #pragma unroll
  for (int i = 0; i < 8; ++i) accC[i] = z8;

  for (int hc = 0; hc < 8; ++hc){
    if (hc < 7){
      __builtin_prefetch(&w1h[(hc+1)*32], 0, 1);            /* global_prefetch_b8 */
      __builtin_prefetch(&w2h[((hc+1)*32)*C_], 0, 1);
    }
    __syncthreads();
    /* stage W1 chunk transposed: Bt[j][k] = W1[k][hc*32+j] */
    for (int idx = tid; idx < 32*128; idx += 256){
      int j = idx & 31;
      int k = idx >> 5;
      Bt[j][k] = w1h[k*HID_ + hc*32 + j];
    }
    /* stage W2 chunk transposed: W2t[nc][k] = W2[hc*32+k][nc] */
    for (int idx = tid; idx < 32*128; idx += 256){
      int nc = idx & 127;
      int k = idx >> 7;
      W2t[nc][k] = w2h[(hc*32 + k)*C_ + nc];
    }
    __syncthreads();

    /* GEMM1: (16x128)x(128x32) per wave */
    v8f a0 = z8, a1 = z8;
    #pragma unroll
    for (int k0 = 0; k0 < 128; k0 += 32){
      v16h af  = load_frag_row(&As[0][0], 128, m0, k0, lane);
      v16h bf0 = load_frag_row(&Bt[0][0], 128, 0,  k0, lane);
      v16h bf1 = load_frag_row(&Bt[0][0], 128, 16, k0, lane);
      a0 = __builtin_amdgcn_wmma_f32_16x16x32_f16(false, af, false, bf0, (short)0, a0, false, false);
      a1 = __builtin_amdgcn_wmma_f32_16x16x32_f16(false, af, false, bf1, (short)0, a1, false, false);
    }
    /* bias + exact GELU -> Hs (wave-private rows, DS ops in-order within wave) */
    #pragma unroll
    for (int r = 0; r < 8; ++r){
      int m = m0 + hf*8 + r;
      float x0 = a0[r] + b1[hc*32 + col];
      float x1 = a1[r] + b1[hc*32 + 16 + col];
      Hs[m][col]      = (_Float16)gelu_f(x0);
      Hs[m][16 + col] = (_Float16)gelu_f(x1);
    }
    /* GEMM2 accumulate: (16x32)x(32x128) per wave */
    v16h ah = load_frag_row(&Hs[0][0], 32, m0, 0, lane);
    #pragma unroll
    for (int nt = 0; nt < 8; ++nt){
      v16h bw = load_frag_row(&W2t[0][0], 32, nt*16, 0, lane);
      accC[nt] = __builtin_amdgcn_wmma_f32_16x16x32_f16(false, ah, false, bw, (short)0, accC[nt], false, false);
    }
  }

  /* epilogue: +b2, channel-first store; 2x b128 stores per tile per lane */
  #pragma unroll
  for (int nt = 0; nt < 8; ++nt){
    int c = nt*16 + col;
    float bias = b2[c];
    float* dst = react + (size_t)(b*C_ + c)*N_ + n0 + m0 + hf*8;
    float4 v0 = { accC[nt][0]+bias, accC[nt][1]+bias, accC[nt][2]+bias, accC[nt][3]+bias };
    float4 v1 = { accC[nt][4]+bias, accC[nt][5]+bias, accC[nt][6]+bias, accC[nt][7]+bias };
    *(float4*)dst       = v0;
    *(float4*)(dst + 4) = v1;
  }
}

/* fused: diffusion stencil + local trilinear + Euler update + optional RMSNorm + reductions */
__global__ void __launch_bounds__(256)
update_kernel(const float* u, const float* react, const float* h,
              const float* lproj, const float* spsks, const float* spskt,
              const float* scal, const float* normw,
              float* ucand, float* unext, float* accum, int do_norm){
  int p = blockIdx.x*blockDim.x + threadIdx.x;   /* 0..BN_-1 */
  int b  = p >> 15;
  int n  = p & (N_ - 1);
  int t  = n >> 12;
  int hh = (n >> 6) & 63;
  int ww = n & 63;

  float dt = scal[0], ag = scal[1], al = scal[2];

  /* trilinear sampling setup from (B,C,2,4,4) local field (half-pixel centers) */
  float st = fminf(fmaxf((t  + 0.5f)*(2.f/T_) - 0.5f, 0.f), 1.f);
  float sh = fminf(fmaxf((hh + 0.5f)*(4.f/H_) - 0.5f, 0.f), 3.f);
  float sw = fminf(fmaxf((ww + 0.5f)*(4.f/W_) - 0.5f, 0.f), 3.f);
  int t0i = (int)st; int t1i = imin_(t0i+1, 1); float ft = st - (float)t0i;
  int h0i = (int)sh; int h1i = imin_(h0i+1, 3); float fh = sh - (float)h0i;
  int w0i = (int)sw; int w1i = imin_(w0i+1, 3); float fw = sw - (float)w0i;

  float sum_du = 0.f, sum_lap = 0.f, sumsq = 0.f;
  size_t bbase = (size_t)b * C_ * N_;

  for (int c = 0; c < C_; ++c){
    const float* uc = u + bbase + (size_t)c * N_;
    float uv = uc[n];

    float diff = 0.f;
    #pragma unroll
    for (int i = 0; i < 3; ++i){
      const int d = (i == 0) ? 1 : ((i == 1) ? 4 : 16);
      float lap = -4.f * uv;
      lap += (hh - d >= 0) ? uc[n - d*64] : 0.f;
      lap += (hh + d < 64) ? uc[n + d*64] : 0.f;
      lap += (ww - d >= 0) ? uc[n - d]    : 0.f;
      lap += (ww + d < 64) ? uc[n + d]    : 0.f;
      diff += spsks[i*C_ + c] * lap;
    }
    #pragma unroll
    for (int i = 0; i < 2; ++i){
      const int d = i + 1;
      float up = (t - d >= 0) ? uc[n - d*4096] : 0.f;
      diff += spskt[i*C_ + c] * (up - uv);
    }
    diff *= DIFF_SCALE;
    sum_lap += fabsf(diff);

    const float* lp = lproj + ((size_t)(b*C_ + c) << 5);
    float c00 = lp[t0i*16 + h0i*4 + w0i]*(1.f-fw) + lp[t0i*16 + h0i*4 + w1i]*fw;
    float c01 = lp[t0i*16 + h1i*4 + w0i]*(1.f-fw) + lp[t0i*16 + h1i*4 + w1i]*fw;
    float c10 = lp[t1i*16 + h0i*4 + w0i]*(1.f-fw) + lp[t1i*16 + h0i*4 + w1i]*fw;
    float c11 = lp[t1i*16 + h1i*4 + w0i]*(1.f-fw) + lp[t1i*16 + h1i*4 + w1i]*fw;
    float lv = (c00*(1.f-fh) + c01*fh)*(1.f-ft) + (c10*(1.f-fh) + c11*fh)*ft;

    float du = diff + react[bbase + (size_t)c*N_ + n] + ag*h[b*C_ + c] + al*lv;
    float cand = uv + dt*du;
    sum_du += fabsf(du);
    sumsq  += cand*cand;
    ucand[bbase + (size_t)c*N_ + n] = cand;
  }

  float scale = do_norm ? rsqrtf(sumsq/(float)C_ + 1e-8f) : 1.f;
  for (int c = 0; c < C_; ++c){
    float cand = ucand[bbase + (size_t)c*N_ + n];   /* same-thread RAW: in order */
    unext[bbase + (size_t)c*N_ + n] = do_norm ? cand*scale*normw[c] : cand;
  }

  float s1 = blockReduceSum(sum_du);
  float s2 = blockReduceSum(sum_lap);
  if (threadIdx.x == 0){
    atomicAdd(&accum[1], s1);
    atomicAdd(&accum[0], s2);
  }
}

__global__ void turb_kernel(const float* cur, const float* prev, float* accum){
  int i = blockIdx.x*blockDim.x + threadIdx.x;
  float a = 0.f, pv = 0.f;
  if (i < PROBE_CELLS){
    float cv = cur[i], pp = prev[i];
    a = fabsf(cv - pp);
    pv = fabsf(pp);
  }
  float s1 = blockReduceSum(a);
  float s2 = blockReduceSum(pv);
  if (threadIdx.x == 0){
    atomicAdd(&accum[2], s1);
    atomicAdd(&accum[3], s2);
  }
}

__global__ void finalize_step_kernel(const float* accum, float* turbs, float* sten, int step){
  if (threadIdx.x == 0 && blockIdx.x == 0){
    const float inv_cells = 1.f/(float)PROBE_CELLS;
    const float inv_bnc   = 1.f/8388608.f;   /* B*N*C */
    float stop_turb = (accum[2]*inv_cells) / (accum[3]*inv_cells + 1e-8f);
    float se = accum[1]*inv_bnc;
    float le = accum[0]*inv_bnc;
    turbs[step] = stop_turb + 0.05f*se + 0.01f*le;
    sten[step]  = se;
  }
}

__global__ void final_out_kernel(const float* turbs, const float* sten, float* out){
  if (threadIdx.x == 0 && blockIdx.x == 0){
    float a = 0.f, bm = 0.f;
    for (int i = 0; i < MAX_STEPS_; ++i){ a += turbs[i]; bm += sten[i]; }
    out[0] = a / (float)MAX_STEPS_;
    out[1] = bm / (float)MAX_STEPS_;
  }
}

/* ---------------- host ---------------- */

extern "C" void kernel_launch(void* const* d_in, const int* in_sizes, int n_in,
                              void* d_out, int out_size, void* d_ws, size_t ws_size,
                              hipStream_t stream){
  (void)in_sizes; (void)n_in; (void)out_size; (void)ws_size;

  const float* u_in  = (const float*)d_in[0];
  const float* w1    = (const float*)d_in[1];
  const float* b1    = (const float*)d_in[2];
  const float* w2    = (const float*)d_in[3];
  const float* b2    = (const float*)d_in[4];
  const float* gw    = (const float*)d_in[5];
  const float* gb    = (const float*)d_in[6];
  const float* vw    = (const float*)d_in[7];
  const float* vb    = (const float*)d_in[8];
  const float* normw = (const float*)d_in[9];
  const float* wlp   = (const float*)d_in[10];
  const float* blp   = (const float*)d_in[11];
  const float* a_g   = (const float*)d_in[12];
  const float* a_l   = (const float*)d_in[13];
  const float* logdt = (const float*)d_in[14];
  const float* ks    = (const float*)d_in[15];
  const float* kt    = (const float*)d_in[16];

  float* Wf = (float*)d_ws;
  size_t o = 0;
  const size_t VOL = (size_t)B_ * C_ * N_;    /* 8388608 */
  float* ubuf0  = Wf + o; o += VOL;
  float* ubuf1  = Wf + o; o += VOL;
  float* ucand  = Wf + o; o += VOL;
  float* react  = Wf + o; o += VOL;
  float* probeA = Wf + o; o += PROBE_CELLS;
  float* probeB = Wf + o; o += PROBE_CELLS;
  float* pooled = Wf + o; o += 256;
  float* hstate = Wf + o; o += 256;
  float* lsmall = Wf + o; o += (size_t)B_*C_*32;
  float* lproj  = Wf + o; o += (size_t)B_*C_*32;
  float* spsks  = Wf + o; o += 3*C_;
  float* spskt  = Wf + o; o += 2*C_;
  float* scal   = Wf + o; o += 8;
  float* accum  = Wf + o; o += 8;
  float* turbs  = Wf + o; o += 16;
  float* sten   = Wf + o; o += 16;
  _Float16* w1h = (_Float16*)(Wf + o); o += (C_*HID_)/2;
  _Float16* w2h = (_Float16*)(Wf + o); o += (HID_*C_)/2;

  float* ubuf[2] = { ubuf0, ubuf1 };

  prep_kernel<<<64, 256, 0, stream>>>(w1, w2, ks, kt, a_g, a_l, logdt,
                                      w1h, w2h, spsks, spskt, scal);
  zero_kernel<<<1, 256, 0, stream>>>(hstate, 256);
  hipMemcpyAsync(ubuf[0], u_in, VOL*sizeof(float), hipMemcpyDeviceToDevice, stream);
  pool_kernel<<<(PROBE_CELLS + 255)/256, 256, 0, stream>>>(ubuf[0], probeA, 2, 8, 8);

  float* pprev = probeA;
  float* pcur  = probeB;

  for (int step = 0; step < MAX_STEPS_; ++step){
    float* ucur  = ubuf[step & 1];
    float* unext = ubuf[(step & 1) ^ 1];
    int do_norm = (((step + 1) % 2) == 0) ? 1 : 0;

    zero_kernel<<<1, 32, 0, stream>>>(accum, 8);
    reaction_mlp_kernel<<<BN_/128, 256, 0, stream>>>(ucur, w1h, b1, w2h, b2, react);
    pooled_mean_kernel<<<B_*C_, 256, 0, stream>>>(ucur, pooled);
    gate_value_kernel<<<1, 256, 0, stream>>>(pooled, gw, gb, vw, vb, hstate);
    pool_kernel<<<(B_*C_*32 + 255)/256, 256, 0, stream>>>(ucur, lsmall, 2, 4, 4);
    local_proj_kernel<<<(B_*C_*32 + 255)/256, 256, 0, stream>>>(lsmall, wlp, blp, lproj);
    update_kernel<<<BN_/256, 256, 0, stream>>>(ucur, react, hstate, lproj, spsks, spskt,
                                               scal, normw, ucand, unext, accum, do_norm);
    pool_kernel<<<(PROBE_CELLS + 255)/256, 256, 0, stream>>>(ucand, pcur, 2, 8, 8);
    turb_kernel<<<PROBE_CELLS/256, 256, 0, stream>>>(pcur, pprev, accum);
    finalize_step_kernel<<<1, 1, 0, stream>>>(accum, turbs, sten, step);

    float* tswap = pprev; pprev = pcur; pcur = tswap;
  }

  hipMemcpyAsync(d_out, ubuf[MAX_STEPS_ & 1], VOL*sizeof(float),
                 hipMemcpyDeviceToDevice, stream);
  final_out_kernel<<<1, 32, 0, stream>>>(turbs, sten, ((float*)d_out) + VOL);
}
